// LocalAttention_6640019439813
// MI455X (gfx1250) — matrix operand order repaired and checked by reading, not compiled, not run on hardware
//
#include <hip/hip_runtime.h>

// ---------------------------------------------------------------------------
// CDNA5 (gfx1250) local+global windowed attention, bf16 WMMA throughout.
// Async global->LDS staging (ASYNCcnt path) used when the toolchain exposes
// the gfx1250 builtins; guarded by __has_builtin with a sync fallback.
// ---------------------------------------------------------------------------

typedef __attribute__((ext_vector_type(16))) __bf16 v16bf;
typedef __attribute__((ext_vector_type(8)))  float  v8f;
typedef int i32x4 __attribute__((vector_size(16)));   // builtin's pointee type

#define NEGMAX (-3.402823466e38f)

#if defined(__has_builtin)
#if __has_builtin(__builtin_amdgcn_global_load_async_to_lds_b128)
#define HAVE_ASYNC_LDS 1
#endif
#endif

__device__ __forceinline__ void wait_async0() {
#if defined(__has_builtin) && __has_builtin(__builtin_amdgcn_s_wait_asynccnt)
  __builtin_amdgcn_s_wait_asynccnt(0);
#else
  asm volatile("s_wait_asynccnt 0x0" ::: "memory");
#endif
}

__device__ __forceinline__ unsigned short f2bf(float f) {
  union { float f; unsigned u; } x; x.f = f;
  unsigned r = x.u + 0x7FFFu + ((x.u >> 16) & 1u);   // round-to-nearest-even
  return (unsigned short)(r >> 16);
}
__device__ __forceinline__ float bf2f(unsigned short b) {
  union { unsigned u; float f; } x; x.u = ((unsigned)b) << 16;
  return x.f;
}

union Frag { v16bf v; uint4 q4[2]; unsigned short u[16]; };

// ---------------------------------------------------------------------------
// Generic bf16 WMMA GEMM: C[M,N] = A[M,K] * B[K,N] (+bias)
// Block tile 128x128, K-chunk 32, 8 waves (each: 16 M-rows x 8 N-tiles).
// A is f32 or bf16 (a_bf16), converted to bf16 in LDS; B is f32 (weights).
// row_mod>0 remaps logical A row r -> r + r/row_mod (for global_tokens[:, :-1]).
// ---------------------------------------------------------------------------
__global__ __launch_bounds__(256) void gemm_wmma(
    const void* __restrict__ Ap, const float* __restrict__ B,
    void* __restrict__ Cp, const float* __restrict__ bias,
    int M, int N, int K, int a_bf16, int c_bf16, int row_mod)
{
  __shared__ unsigned short As [128 * 32];  // [m][k]
  __shared__ unsigned short Bts[128 * 32];  // [n][k]  (B transposed)
  const int tid  = threadIdx.x;
  const int wt   = tid >> 5;     // wave 0..7 -> 16-row M strip
  const int lane = tid & 31;
  const int hi   = lane >> 4;    // lane half (0/1)
  const int ln   = lane & 15;
  const int m0   = blockIdx.y * 128;
  const int n0   = blockIdx.x * 128;
  const float*          A32 = (const float*)Ap;
  const unsigned short* A16 = (const unsigned short*)Ap;

  v8f acc[8];
#pragma unroll
  for (int i = 0; i < 8; ++i) acc[i] = (v8f){0,0,0,0,0,0,0,0};

  for (int k0 = 0; k0 < K; k0 += 32) {
    // stage A 128x32 (convert f32 -> bf16 on the fly)
    for (int i = tid; i < 128 * 32; i += 256) {
      int m = i >> 5, kk = i & 31;
      int r  = m0 + m;
      int pr = r + (row_mod ? (r / row_mod) : 0);
      As[m * 32 + kk] = a_bf16 ? A16[(long)pr * K + k0 + kk]
                               : f2bf(A32[(long)pr * K + k0 + kk]);
    }
    // stage B^T 128x32 (coalesced reads over N)
    for (int i = tid; i < 128 * 32; i += 256) {
      int kk = i >> 7, n = i & 127;
      Bts[n * 32 + kk] = f2bf(B[(long)(k0 + kk) * N + n0 + n]);
    }
    if (k0 + 32 < K)  // warm GL2 for the next B chunk -> global_prefetch_b8
      __builtin_prefetch(&B[(long)(k0 + 32) * N + n0 + (tid & 127)], 0, 1);
    __syncthreads();

    // A fragment: 16x32, layout: halves 0..7 -> K = 8*hi + 0..7,
    //                            halves 8..15 -> K = 16 + 8*hi + 0..7
    Frag a;
    {
      const unsigned short* ap = &As[(16 * wt + ln) * 32];
      a.q4[0] = *(const uint4*)&ap[8 * hi];
      a.q4[1] = *(const uint4*)&ap[16 + 8 * hi];
    }
#pragma unroll
    for (int nt = 0; nt < 8; ++nt) {
      // B fragment: 32x16, lane ln = column, K = 8*hi + 0..7, 16 + 8*hi + 0..7
      Frag b;
      const unsigned short* bp = &Bts[(16 * nt + ln) * 32];
      b.q4[0] = *(const uint4*)&bp[8 * hi];
      b.q4[1] = *(const uint4*)&bp[16 + 8 * hi];
      acc[nt] = __builtin_amdgcn_wmma_f32_16x16x32_bf16(
          false, a.v, false, b.v, (short)0, acc[nt], false, false);
    }
    __syncthreads();
  }

  float*          C32 = (float*)Cp;
  unsigned short* C16 = (unsigned short*)Cp;
#pragma unroll
  for (int nt = 0; nt < 8; ++nt) {
    int col = n0 + 16 * nt + ln;
#pragma unroll
    for (int r = 0; r < 8; ++r) {
      int row = m0 + 16 * wt + r + 8 * hi;   // C layout: M = r + 8*hi
      float v = acc[nt][r];
      if (c_bf16) C16[(long)row * N + col] = f2bf(v);
      else        C32[(long)row * N + col] = v + (bias ? bias[col] : 0.0f);
    }
  }
}

// ---------------------------------------------------------------------------
// RoPE + head-split scatter. Also folds softmax scale (1/8) into Q.
// q_tmp [16384][1024] bf16, kv_tmp [16384][2048] bf16 (k = first 1024 cols).
// Outputs q_bh/k_bh as [b*h][4096][64] bf16.
// ---------------------------------------------------------------------------
__global__ __launch_bounds__(256) void rope_scatter(
    const unsigned short* __restrict__ q_tmp,
    const unsigned short* __restrict__ kv_tmp,
    const float* __restrict__ pos_emb,
    unsigned short* __restrict__ q_bh,
    unsigned short* __restrict__ k_bh)
{
  int gid  = blockIdx.x * 256 + threadIdx.x;   // (row, head)
  int row  = gid >> 4;
  int head = gid & 15;
  int b    = row >> 12;
  int pos  = row & 4095;
  int bh   = b * 16 + head;
  const unsigned short* qp = q_tmp  + (long)row * 1024 + head * 64;
  const unsigned short* kp = kv_tmp + (long)row * 2048 + head * 64;
  unsigned short* qo = q_bh + ((long)bh * 4096 + pos) * 64;
  unsigned short* ko = k_bh + ((long)bh * 4096 + pos) * 64;
  const float* pe = pos_emb + pos * 64;
#pragma unroll
  for (int d = 0; d < 32; ++d) {
    float c0 = __cosf(pe[d]),      s0 = __sinf(pe[d]);
    float c1 = __cosf(pe[d + 32]), s1 = __sinf(pe[d + 32]);
    float q0 = bf2f(qp[d]), q1 = bf2f(qp[d + 32]);
    float k0 = bf2f(kp[d]), k1 = bf2f(kp[d + 32]);
    // rotate_half: out[d] = x[d]*cos - x[d+32]*sin; out[d+32] = x[d+32]*cos + x[d]*sin
    qo[d]      = f2bf((q0 * c0 - q1 * s0) * 0.125f);
    qo[d + 32] = f2bf((q1 * c1 + q0 * s1) * 0.125f);
    ko[d]      = f2bf(k0 * c0 - k1 * s0);
    ko[d + 32] = f2bf(k1 * c1 + k0 * s1);
  }
}

// gkv_tmp [256][2048] -> gk/gv [b*h][64][64] bf16 (no RoPE on globals).
__global__ __launch_bounds__(256) void gkv_scatter(
    const unsigned short* __restrict__ gkv,
    unsigned short* __restrict__ gk,
    unsigned short* __restrict__ gv)
{
  int gid  = blockIdx.x * 256 + threadIdx.x;   // 262144 total
  int dd   = gid & 63;
  int tok  = (gid >> 6) & 63;
  int head = (gid >> 12) & 15;
  int b    = gid >> 16;
  long src = (long)(b * 64 + tok) * 2048 + head * 64 + dd;
  long dst = (((long)(b * 16 + head)) * 64 + tok) * 64 + dd;
  gk[dst] = gkv[src];
  gv[dst] = gkv[src + 1024];
}

// ---------------------------------------------------------------------------
// Fused windowed attention. grid = (nw=32, b*h=64), 256 threads = 8 waves.
// Keys: 64 global + 256 local (prev+curr window, zero-padded for window 0).
// Reference quirk: local values == rotated local keys (vw2 = kw2), so V^T is
// built from the K tile already resident in LDS (no second global read).
// LDS: K[320][64] bf16 (40KB) + V^T[64][320] (40KB) + S/P[128][320] (80KB).
// ---------------------------------------------------------------------------
__global__ __launch_bounds__(256) void local_attn(
    const unsigned short* __restrict__ q_bh,
    const unsigned short* __restrict__ k_bh,
    const unsigned short* __restrict__ gk,
    const unsigned short* __restrict__ gv,
    unsigned short* __restrict__ attn_out)     // [16384][1024] bf16
{
  extern __shared__ unsigned short smem[];
  unsigned short* Kls = smem;                  // [320][64]
  unsigned short* Vt  = smem + 320 * 64;       // [64][320]
  unsigned short* S   = Vt   + 64 * 320;       // [128][320]
  const int wi   = blockIdx.x;
  const int bh   = blockIdx.y;
  const int b    = bh >> 4;
  const int head = bh & 15;
  const int tid  = threadIdx.x;
  const int wt   = tid >> 5;
  const int lane = tid & 31;
  const int hi   = lane >> 4;
  const int ln   = lane & 15;

  // ---- stage K rows (global tokens + 256 contiguous local rows) ----
  // global-token rows (j < 64): gather normally (64x64 elems)
  for (int i = tid; i < 64 * 64; i += 256) {
    int j = i >> 6, dd = i & 63;
    Kls[j * 64 + dd] = gk[((long)bh * 64 + j) * 64 + dd];
  }
#if HAVE_ASYNC_LDS
  // local rows: contiguous 32KB of k_bh -> LDS via async DMA (ASYNCcnt),
  // no VGPR round-trip. 2048 x b128 transfers, 8 per thread.
  for (int c = tid; c < 2048; c += 256) {
    int jj = c >> 3, cc = c & 7;                 // row 0..255, 16B chunk 0..7
    int p  = (wi - 1) * 128 + jj;                // source position
    unsigned short* ldst = &Kls[(64 + jj) * 64 + cc * 8];
    if (p >= 0) {
      const unsigned short* gsrc =
          k_bh + ((long)bh * 4096 + p) * 64 + cc * 8;
      __builtin_amdgcn_global_load_async_to_lds_b128(
          (__attribute__((address_space(1))) i32x4*)gsrc,
          (__attribute__((address_space(3))) i32x4*)ldst, 0, 0);
    } else {
      *(uint4*)ldst = (uint4){0, 0, 0, 0};       // window-0 zero pad
    }
  }
  wait_async0();
#else
  for (int i = tid; i < 256 * 64; i += 256) {
    int jj = i >> 6, dd = i & 63;
    int p = (wi - 1) * 128 + jj;
    Kls[(64 + jj) * 64 + dd] =
        (p >= 0) ? k_bh[((long)bh * 4096 + p) * 64 + dd] : (unsigned short)0;
  }
#endif
  __syncthreads();

  // ---- build V^T: global part from gv, local part from K already in LDS ----
  for (int i = tid; i < 320 * 64; i += 256) {
    int j = i >> 6, dd = i & 63;
    unsigned short vv = (j < 64) ? gv[((long)bh * 64 + j) * 64 + dd]
                                 : Kls[j * 64 + dd];   // vw2 = kw2
    Vt[dd * 320 + j] = vv;
  }
  __syncthreads();

  // ---- per-wave Q fragments (16 rows x K=64 -> two 16x32 A operands) ----
  Frag a0, a1;
  {
    const unsigned short* qp =
        q_bh + ((long)bh * 4096 + wi * 128 + 16 * wt + ln) * 64;
    a0.q4[0] = *(const uint4*)&qp[8 * hi];
    a0.q4[1] = *(const uint4*)&qp[16 + 8 * hi];
    a1.q4[0] = *(const uint4*)&qp[32 + 8 * hi];
    a1.q4[1] = *(const uint4*)&qp[48 + 8 * hi];
  }

  // ---- S = (Q * scale) K^T, masked, stored bf16 ----
#pragma unroll 4
  for (int jt = 0; jt < 20; ++jt) {
    v8f c = (v8f){0,0,0,0,0,0,0,0};
    Frag b0, b1;
    const unsigned short* kp = &Kls[(16 * jt + ln) * 64];
    b0.q4[0] = *(const uint4*)&kp[8 * hi];
    b0.q4[1] = *(const uint4*)&kp[16 + 8 * hi];
    b1.q4[0] = *(const uint4*)&kp[32 + 8 * hi];
    b1.q4[1] = *(const uint4*)&kp[48 + 8 * hi];
    c = __builtin_amdgcn_wmma_f32_16x16x32_bf16(false, a0.v, false, b0.v, (short)0, c, false, false);
    c = __builtin_amdgcn_wmma_f32_16x16x32_bf16(false, a1.v, false, b1.v, (short)0, c, false, false);
    int j = 16 * jt + ln;
#pragma unroll
    for (int r = 0; r < 8; ++r) {
      int irow = 16 * wt + r + 8 * hi;
      // global mask: j >= wi+33 ; local causal: (j-64) >= irow+129
      bool msk = (j < 64) ? (j >= wi + 33) : ((j - 64) >= irow + 129);
      S[irow * 320 + j] = f2bf(msk ? NEGMAX : c[r]);
    }
  }
  __syncthreads();

  // ---- softmax: one thread per row (128 rows) ----
  if (tid < 128) {
    unsigned short* sp = &S[tid * 320];
    float mx = -3.4e38f;
    for (int j = 0; j < 320; ++j) mx = fmaxf(mx, bf2f(sp[j]));
    float sum = 0.f;
    for (int j = 0; j < 320; ++j) {
      float e = __expf(bf2f(sp[j]) - mx);
      sum += e;
      sp[j] = f2bf(e);
    }
    float inv = 1.0f / sum;
    for (int j = 0; j < 320; ++j) sp[j] = f2bf(bf2f(sp[j]) * inv);
  }
  __syncthreads();

  // ---- O = P V : 4 output N-tiles x 10 K-chunks of 32 ----
  long orow_base = (long)b * 4096 + wi * 128;
#pragma unroll
  for (int nt = 0; nt < 4; ++nt) {
    v8f o = (v8f){0,0,0,0,0,0,0,0};
#pragma unroll 2
    for (int kc = 0; kc < 10; ++kc) {
      int k0 = 32 * kc;
      Frag pa, vb;
      const unsigned short* pp = &S[(16 * wt + ln) * 320 + k0];
      pa.q4[0] = *(const uint4*)&pp[8 * hi];
      pa.q4[1] = *(const uint4*)&pp[16 + 8 * hi];
      const unsigned short* vp = &Vt[(16 * nt + ln) * 320 + k0];
      vb.q4[0] = *(const uint4*)&vp[8 * hi];
      vb.q4[1] = *(const uint4*)&vp[16 + 8 * hi];
      o = __builtin_amdgcn_wmma_f32_16x16x32_bf16(false, pa.v, false, vb.v, (short)0, o, false, false);
    }
#pragma unroll
    for (int r = 0; r < 8; ++r) {
      int irow = 16 * wt + r + 8 * hi;
      attn_out[(orow_base + irow) * 1024 + head * 64 + 16 * nt + ln] = f2bf(o[r]);
    }
  }
}

// ---------------------------------------------------------------------------
// Host orchestration (all on `stream`, scratch from d_ws only).
// ---------------------------------------------------------------------------
extern "C" void kernel_launch(void* const* d_in, const int* in_sizes, int n_in,
                              void* d_out, int out_size, void* d_ws, size_t ws_size,
                              hipStream_t stream) {
  const float* x    = (const float*)d_in[0];   // [4,4096,1024]
  const float* gt   = (const float*)d_in[1];   // [4,65,1024]
  const float* pe   = (const float*)d_in[2];   // [4096,64]
  const float* Wq   = (const float*)d_in[3];   // [1024,1024]
  const float* Wkv  = (const float*)d_in[4];   // [1024,2048]
  const float* Wout = (const float*)d_in[5];   // [1024,1024]
  const float* bout = (const float*)d_in[6];   // [1024]
  (void)in_sizes; (void)n_in; (void)out_size; (void)ws_size;

  char* ws = (char*)d_ws;
  size_t off = 0;
  auto alloc = [&](size_t bytes) {
    void* p = ws + off; off += (bytes + 255) & ~(size_t)255; return p;
  };
  unsigned short* q_tmp  = (unsigned short*)alloc((size_t)16384 * 1024 * 2); // reused as attn_out
  unsigned short* kv_tmp = (unsigned short*)alloc((size_t)16384 * 2048 * 2);
  unsigned short* gkv    = (unsigned short*)alloc((size_t)256   * 2048 * 2);
  unsigned short* q_bh   = (unsigned short*)alloc((size_t)64 * 4096 * 64 * 2);
  unsigned short* k_bh   = (unsigned short*)alloc((size_t)64 * 4096 * 64 * 2);
  unsigned short* gkb    = (unsigned short*)alloc((size_t)64 * 64 * 64 * 2);
  unsigned short* gvb    = (unsigned short*)alloc((size_t)64 * 64 * 64 * 2);

  dim3 blk(256);
  // q = x @ Wq               (bf16 out)
  gemm_wmma<<<dim3(8, 128),  blk, 0, stream>>>(x,  Wq,  q_tmp,  nullptr, 16384, 1024, 1024, 0, 1, 0);
  // kv = x @ Wkv             (bf16 out)
  gemm_wmma<<<dim3(16, 128), blk, 0, stream>>>(x,  Wkv, kv_tmp, nullptr, 16384, 2048, 1024, 0, 1, 0);
  // gkv = global_tokens[:, :-1] @ Wkv  (row remap r -> r + r/64)
  gemm_wmma<<<dim3(16, 2),   blk, 0, stream>>>(gt, Wkv, gkv,    nullptr,   256, 2048, 1024, 0, 1, 64);
  // RoPE + head split (+ fold 1/sqrt(64) into q)
  rope_scatter<<<dim3(1024), blk, 0, stream>>>(q_tmp, kv_tmp, pe, q_bh, k_bh);
  gkv_scatter<<<dim3(1024),  blk, 0, stream>>>(gkv, gkb, gvb);
  // fused attention (attn output aliases q_tmp, which is dead by now)
  local_attn<<<dim3(32, 64), blk, 163840, stream>>>(q_bh, k_bh, gkb, gvb, q_tmp);
  // out = attn @ Wout + bout (fp32 out, bf16 A)
  gemm_wmma<<<dim3(8, 128),  blk, 0, stream>>>(q_tmp, Wout, d_out, bout, 16384, 1024, 1024, 1, 0, 0);
}
